// Model_558345749103
// MI455X (gfx1250) — matrix operand order repaired and checked by reading, not compiled, not run on hardware
//
#include <hip/hip_runtime.h>
#include <hip/hip_bf16.h>
#include <math.h>

#define N_HEADS   8
#define D_MODEL   512
#define HEAD_SIZE 64
#define VOCAB     16384
#define BLOCK_T   512
#define BATCH     4
#define ROWS_PER_HEAD (BATCH * BLOCK_T)          // 2048 query rows per head
#define QTILE     32                              // queries per wave (2 groups of 16)
#define QT_PER_HEAD (ROWS_PER_HEAD / QTILE)       // 64
#define KSPLIT    8                               // flash-decoding key splits
#define KEYS_PER_SPLIT (VOCAB / KSPLIT)           // 2048
#define KB        32                              // keys per inner iteration
#define NTILE16   (N_HEADS * QT_PER_HEAD * 2)     // 1024 16-query tiles

typedef __attribute__((ext_vector_type(16))) __bf16 v16bf;
typedef __attribute__((ext_vector_type(8)))  float  v8f;

// fp32 -> bf16 bits RNE for the prep kernels (bulk converts, not hot path)
static __device__ __forceinline__ unsigned short f2bf(float f) {
  unsigned u = __float_as_uint(f);
  u += 0x7fffu + ((u >> 16) & 1u);
  return (unsigned short)(u >> 16);
}

// Pack two fp32 into one u32 holding {bf16(a) | bf16(b)<<16}; prefer the
// single-instruction v_cvt_pk_bf16_f32 path when the builtin is declared.
#if defined(__has_builtin)
#if __has_builtin(__builtin_amdgcn_cvt_pk_bf16_f32)
#define HAVE_PK_BF16 1
#endif
#endif

static __device__ __forceinline__ unsigned pk_bf16(float a, float b) {
#ifdef HAVE_PK_BF16
  typedef __attribute__((ext_vector_type(2))) __bf16 v2bf;
  const v2bf p = __builtin_amdgcn_cvt_pk_bf16_f32(a, b);
  return __builtin_bit_cast(unsigned, p);
#else
  union { __bf16 b[2]; unsigned u; } t;
  t.b[0] = (__bf16)a;
  t.b[1] = (__bf16)b;
  return t.u;
#endif
}

// Assemble a 16-half WMMA operand from two 16B-aligned global chunks (2x b128 loads)
static __device__ __forceinline__ v16bf load2(const unsigned short* p0,
                                              const unsigned short* p1) {
  union { uint4 u[2]; v16bf v; } t;
  t.u[0] = *(const uint4*)p0;
  t.u[1] = *(const uint4*)p1;
  return t.v;
}

static __device__ __forceinline__ v8f wmma_bf16(v16bf a, v16bf b, v8f c) {
  return __builtin_amdgcn_wmma_f32_16x16x32_bf16(false, a, false, b, (short)0, c,
                                                 false, false);
}

// ---------------------------------------------------------------------------
// Kernel 1: encoder + sinusoidal PE + LayerNorm -> Q (bf16).
// Folds BOTH the 1/sqrt(512) score scale and log2(e) into Q so that the
// attention kernel can use raw v_exp_f32 with no per-element multiplies.
// One wave per (t, b, h); lane handles hs = 2*lane, 2*lane+1 (sin/cos pair).
// ---------------------------------------------------------------------------
__global__ void __launch_bounds__(32)
qprep_kernel(const float* __restrict__ x, const float* __restrict__ enc_w,
             const float* __restrict__ enc_b, const float* __restrict__ ln_w,
             const float* __restrict__ ln_b, unsigned short* __restrict__ Qb) {
  const int t = blockIdx.x, b = blockIdx.y, h = blockIdx.z;
  const int lane = threadIdx.x;
  const int hs0 = lane * 2, hs1 = hs0 + 1;
  const float xv = x[b * BLOCK_T + t];
  const int d0 = h * HEAD_SIZE + hs0;
  float v0 = xv * enc_w[d0] + enc_b[d0];
  float v1 = xv * enc_w[d0 + 1] + enc_b[d0 + 1];
  // pe: div = exp(-(ln 1e4) * (hs&~1) / 64); even->sin, odd->cos
  const float dv = __expf(-9.2103403719761836f * (float)hs0 * (1.0f / 64.0f));
  const float arg = (float)t * dv;
  v0 += __sinf(arg);
  v1 += __cosf(arg);
  float s = v0 + v1;
  #pragma unroll
  for (int off = 16; off; off >>= 1) s += __shfl_xor(s, off);
  const float mu = s * (1.0f / 64.0f);
  const float e0 = v0 - mu, e1 = v1 - mu;
  float vs = e0 * e0 + e1 * e1;
  #pragma unroll
  for (int off = 16; off; off >>= 1) vs += __shfl_xor(vs, off);
  const float inv = rsqrtf(vs * (1.0f / 64.0f) + 1e-5f);
  // 1/sqrt(512) * log2(e): scores come out of WMMA already in log2 units
  const float SC = 0.04419417382415922f * 1.4426950408889634f;
  const float q0 = (e0 * inv * ln_w[hs0] + ln_b[hs0]) * SC;
  const float q1 = (e1 * inv * ln_w[hs1] + ln_b[hs1]) * SC;
  const size_t ro = ((size_t)h * ROWS_PER_HEAD + (size_t)b * BLOCK_T + t) * HEAD_SIZE;
  Qb[ro + hs0] = f2bf(q0);
  Qb[ro + hs1] = f2bf(q1);
}

// ---------------------------------------------------------------------------
// Kernel 2a/2b: embedding fp32 -> bf16, row-major (Eb) and transposed (Ev)
// ---------------------------------------------------------------------------
__global__ void ebprep_kernel(const float* __restrict__ E, unsigned short* __restrict__ Eb) {
  const size_t i = (size_t)blockIdx.x * blockDim.x + threadIdx.x;
  if (i < (size_t)N_HEADS * VOCAB * HEAD_SIZE) Eb[i] = f2bf(E[i]);
}

__global__ void evprep_kernel(const float* __restrict__ E, unsigned short* __restrict__ Ev) {
  const size_t i = (size_t)blockIdx.x * blockDim.x + threadIdx.x;
  if (i >= (size_t)N_HEADS * VOCAB * HEAD_SIZE) return;
  const int v = (int)(i & (VOCAB - 1));          // fastest: coalesced writes
  const int d = (int)((i >> 14) & (HEAD_SIZE - 1));
  const int h = (int)(i >> 20);
  Ev[i] = f2bf(E[((size_t)h * VOCAB + v) * HEAD_SIZE + d]);  // Ev[h][d][v] = E[h][v][d]
}

// ---------------------------------------------------------------------------
// Kernel 3: flash attention, one wave per (query tile of 32, head, key split).
// S^T = E_blk x Q^T (A = E rows, B = Q rows, both direct b128 loads).
// Softmax WITHOUT max subtraction: |score| is bounded (~4) because Q is
// LayerNormed and pre-scaled by 1/sqrt(512), so exp2 never overflows/denorms
// and the result is mathematically identical; raw v_exp_f32 is exact here.
// Then O^T += V^T x P^T with A = Ev rows and B = P^T assembled from the S^T
// C-tiles via packed-u32 shfl_xor(16).
// ---------------------------------------------------------------------------
__global__ void __launch_bounds__(32)
attn_kernel(const unsigned short* __restrict__ Qb,
            const unsigned short* __restrict__ Eb,
            const unsigned short* __restrict__ Ev,
            float* __restrict__ Ows, float* __restrict__ lws) {
  const int qt = blockIdx.x, h = blockIdx.y, split = blockIdx.z;
  const int lane = threadIdx.x;
  const int hf = lane >> 4, ln = lane & 15;

  const unsigned short* EbH = Eb + (size_t)h * VOCAB * HEAD_SIZE;
  const unsigned short* EvH = Ev + (size_t)h * HEAD_SIZE * VOCAB;

  // Q as B operand (K=hs, N=queries): lane ln = query, halves = 16 hs values.
  v16bf bq[2][2];
  #pragma unroll
  for (int g = 0; g < 2; ++g) {
    const unsigned short* qr =
        Qb + ((size_t)h * ROWS_PER_HEAD + qt * QTILE + g * 16 + ln) * HEAD_SIZE;
    bq[g][0] = load2(qr + hf * 8, qr + 16 + hf * 8);        // hs [0,32)
    bq[g][1] = load2(qr + 32 + hf * 8, qr + 48 + hf * 8);  // hs [32,64)
  }

  const v8f vzero = {0.f, 0.f, 0.f, 0.f, 0.f, 0.f, 0.f, 0.f};
  v8f o[2][4];  // O^T accumulators: 64 hs x 16 queries per group
  #pragma unroll
  for (int g = 0; g < 2; ++g)
    #pragma unroll
    for (int nt = 0; nt < 4; ++nt) o[g][nt] = vzero;

  float l[2] = {0.f, 0.f};  // running sum of exp(score) per query (lane)

  const int k0 = split * KEYS_PER_SPLIT;
  for (int kb = k0; kb < k0 + KEYS_PER_SPLIT; kb += KB) {
    // E rows as A operand (M=keys, K=hs). lane key = kb(+16) + ln.
    const unsigned short* er0 = EbH + (size_t)(kb + ln) * HEAD_SIZE;
    const unsigned short* er1 = er0 + 16 * HEAD_SIZE;
    const v16bf ae00 = load2(er0 + hf * 8, er0 + 16 + hf * 8);       // keys 0-15, hs 0-31
    const v16bf ae01 = load2(er0 + 32 + hf * 8, er0 + 48 + hf * 8);  // keys 0-15, hs 32-63
    const v16bf ae10 = load2(er1 + hf * 8, er1 + 16 + hf * 8);       // keys 16-31
    const v16bf ae11 = load2(er1 + 32 + hf * 8, er1 + 48 + hf * 8);

    v16bf bp[2];
    #pragma unroll
    for (int g = 0; g < 2; ++g) {
      v8f s0 = vzero, s1 = vzero;  // S^T C-tiles: lane = query, vgpr r = key r+8*hf
      s0 = wmma_bf16(ae00, bq[g][0], s0);
      s0 = wmma_bf16(ae01, bq[g][1], s0);
      s1 = wmma_bf16(ae10, bq[g][0], s1);
      s1 = wmma_bf16(ae11, bq[g][1], s1);

      // p = exp2(s) (log2 scale pre-folded into Q); raw v_exp_f32, no guard.
      // Convert directly into packed bf16 pairs (one v_cvt_pk_bf16_f32 each).
      unsigned pu0[4], pu1[4];
      float rs = 0.f;
      #pragma unroll
      for (int j = 0; j < 4; ++j) {
        const float a0 = __builtin_amdgcn_exp2f(s0[2 * j]);
        const float a1 = __builtin_amdgcn_exp2f(s0[2 * j + 1]);
        const float b0 = __builtin_amdgcn_exp2f(s1[2 * j]);
        const float b1 = __builtin_amdgcn_exp2f(s1[2 * j + 1]);
        rs += (a0 + a1) + (b0 + b1);
        pu0[j] = pk_bf16(a0, a1);  // keys kb + 8*hf + 2j..2j+1      (tile 0)
        pu1[j] = pk_bf16(b0, b1);  // keys kb + 16 + 8*hf + 2j..2j+1 (tile 1)
      }
      l[g] += rs + __shfl_xor(rs, 16);

      // Build P^T as B operand (K=keys, N=queries) from the two C-tiles,
      // exchanging packed bf16 pairs across lane halves:
      // low halves  = keys kb+16*hf+0..7 : own pu0 (hf=0) / partner pu1 (hf=1)
      // high halves = keys kb+16*hf+8..15: partner pu0 (hf=0) / own pu1 (hf=1)
      union { unsigned u[8]; v16bf v; } ub;
      #pragma unroll
      for (int j = 0; j < 4; ++j) {
        const unsigned xa = __shfl_xor(pu0[j], 16);
        const unsigned xb = __shfl_xor(pu1[j], 16);
        ub.u[j]     = hf ? xb : pu0[j];
        ub.u[4 + j] = hf ? pu1[j] : xa;
      }
      bp[g] = ub.v;
    }

    // O^T += V^T x P^T ; A = Ev rows (M=hs, K=keys), shared across both groups
    #pragma unroll
    for (int nt = 0; nt < 4; ++nt) {
      const unsigned short* vr = EvH + (size_t)(nt * 16 + ln) * VOCAB + kb;
      const v16bf av = load2(vr + hf * 16, vr + hf * 16 + 8);
      o[0][nt] = wmma_bf16(av, bp[0], o[0][nt]);
      o[1][nt] = wmma_bf16(av, bp[1], o[1][nt]);
    }
  }

  // write split partials: per lane 32 O floats + l
  #pragma unroll
  for (int g = 0; g < 2; ++g) {
    const int tile16 = (h * QT_PER_HEAD + qt) * 2 + g;
    const size_t base = ((size_t)tile16 * KSPLIT + split) * 32 + lane;
    lws[base] = l[g];
    float* op = Ows + base * 32;
    #pragma unroll
    for (int nt = 0; nt < 4; ++nt)
      #pragma unroll
      for (int r = 0; r < 8; ++r) op[nt * 8 + r] = o[g][nt][r];
  }
}

// ---------------------------------------------------------------------------
// Kernel 4: merge key splits (plain sums; no max tracking needed) + decoder
// dot -> per-(row,head) partials (deterministic, no atomics).
// ---------------------------------------------------------------------------
__global__ void __launch_bounds__(32)
reduce_kernel(const float* __restrict__ Ows, const float* __restrict__ lws,
              const float* __restrict__ dec_w, float* __restrict__ pacc) {
  const int tile16 = blockIdx.x;
  const int lane = threadIdx.x, hf = lane >> 4, ln = lane & 15;
  const int h = tile16 >> 7, rem = tile16 & 127, qt = rem >> 1, g = rem & 1;

  float L = 0.f, acc[32];
  #pragma unroll
  for (int v = 0; v < 32; ++v) acc[v] = 0.f;
  #pragma unroll
  for (int s = 0; s < KSPLIT; ++s) {
    const size_t base = ((size_t)tile16 * KSPLIT + s) * 32 + lane;
    L += lws[base];
    const float* op = Ows + base * 32;
    #pragma unroll
    for (int v = 0; v < 32; ++v) acc[v] += op[v];
  }
  const float inv = 1.0f / L;
  float partial = 0.f;
  #pragma unroll
  for (int v = 0; v < 32; ++v) {
    const int hs = (v >> 3) * 16 + (v & 7) + 8 * hf;  // O^T C-layout mapping
    partial += acc[v] * inv * dec_w[h * HEAD_SIZE + hs];
  }
  partial += __shfl_xor(partial, 16);
  if (hf == 0) {
    const int row = qt * QTILE + g * 16 + ln;  // = b*512 + t
    pacc[(size_t)row * N_HEADS + h] = partial;
  }
}

// Kernel 5: out[row] = dec_b + sum over heads
__global__ void final_kernel(const float* __restrict__ pacc,
                             const float* __restrict__ dec_b,
                             float* __restrict__ out) {
  const int i = blockIdx.x * blockDim.x + threadIdx.x;
  if (i < BATCH * BLOCK_T) {
    float s = dec_b[0];
    #pragma unroll
    for (int h = 0; h < N_HEADS; ++h) s += pacc[(size_t)i * N_HEADS + h];
    out[i] = s;
  }
}

extern "C" void kernel_launch(void* const* d_in, const int* in_sizes, int n_in,
                              void* d_out, int out_size, void* d_ws, size_t ws_size,
                              hipStream_t stream) {
  (void)in_sizes; (void)n_in; (void)out_size; (void)ws_size;
  const float* x     = (const float*)d_in[0];
  const float* emb   = (const float*)d_in[1];
  const float* enc_w = (const float*)d_in[2];
  const float* enc_b = (const float*)d_in[3];
  const float* ln_w  = (const float*)d_in[4];
  const float* ln_b  = (const float*)d_in[5];
  const float* dec_w = (const float*)d_in[6];
  const float* dec_b = (const float*)d_in[7];
  float* out = (float*)d_out;

  // workspace carve (~67 MB total), all regions power-of-two aligned
  char* w = (char*)d_ws;
  const size_t QB_BYTES  = (size_t)N_HEADS * ROWS_PER_HEAD * HEAD_SIZE * 2;  //  2 MB
  const size_t E_BYTES   = (size_t)N_HEADS * VOCAB * HEAD_SIZE * 2;          // 16 MB
  const size_t OWS_BYTES = (size_t)NTILE16 * KSPLIT * 32 * 32 * 4;           // 32 MB
  const size_t L_BYTES   = (size_t)NTILE16 * KSPLIT * 32 * 4;                //  1 MB
  unsigned short* Qb = (unsigned short*)w;  w += QB_BYTES;
  unsigned short* Eb = (unsigned short*)w;  w += E_BYTES;
  unsigned short* Ev = (unsigned short*)w;  w += E_BYTES;
  float* Ows = (float*)w;                   w += OWS_BYTES;
  float* lws = (float*)w;                   w += L_BYTES;
  float* pacc = (float*)w;                  // 2048*8 floats = 64 KB

  qprep_kernel<<<dim3(BLOCK_T, BATCH, N_HEADS), 32, 0, stream>>>(x, enc_w, enc_b,
                                                                 ln_w, ln_b, Qb);
  const size_t NE = (size_t)N_HEADS * VOCAB * HEAD_SIZE;  // 8,388,608
  ebprep_kernel<<<dim3((unsigned)((NE + 255) / 256)), 256, 0, stream>>>(emb, Eb);
  evprep_kernel<<<dim3((unsigned)((NE + 255) / 256)), 256, 0, stream>>>(emb, Ev);
  attn_kernel<<<dim3(QT_PER_HEAD, N_HEADS, KSPLIT), 32, 0, stream>>>(Qb, Eb, Ev,
                                                                     Ows, lws);
  reduce_kernel<<<dim3(NTILE16), 32, 0, stream>>>(Ows, lws, dec_w, pacc);
  final_kernel<<<dim3((BATCH * BLOCK_T + 255) / 256), 256, 0, stream>>>(pacc, dec_b, out);
}
